// DGLMPN_56221121904822
// MI455X (gfx1250) — matrix operand order repaired and checked by reading, not compiled, not run on hardware
//
#include <hip/hip_runtime.h>
#include <hip/hip_bf16.h>

#define H        256
#define AF       39
#define BFD      11
#define NMOLS    10000
#define APM      20
#define NNODES   (NMOLS * APM)          // 200000
#define NEDGES   (2 * NMOLS * (APM-1))  // 380000
#define DEPTH    6

typedef __attribute__((ext_vector_type(16))) __bf16 v16bf;
typedef __attribute__((ext_vector_type(8)))  __bf16 v8bf;
typedef __attribute__((ext_vector_type(8)))  float  v8f;

// ---------------------------------------------------------------------------
// Weight packing: W [256 x Kreal] row-major f32 -> bf16 fragments laid out so
// each lane's 32B B-fragment (V_WMMA_F32_16X16X32_BF16 B operand) is contiguous.
// B 32x16 (KxN) layout: lanes 0-15: N=lane, K=kk*32+{0..7,16..23} ; lanes 16-31:
// N=lane-16, K=kk*32+{8..15,24..31}.  index = ((nt*nkk + kk)*32 + lane)*16 + i
// ---------------------------------------------------------------------------
__global__ __launch_bounds__(256) void mpnn_pack_w(
    const float* __restrict__ W, int Kreal, int nkk, __bf16* __restrict__ out) {
  int idx = blockIdx.x * blockDim.x + threadIdx.x;
  int total = 16 * nkk * 32 * 16;
  if (idx >= total) return;
  int i    = idx & 15;
  int lane = (idx >> 4) & 31;
  int tile = idx >> 9;
  int kk   = tile % nkk;
  int nt   = tile / nkk;
  int n = nt * 16 + (lane & 15);
  int k = kk * 32 + ((lane < 16) ? 0 : 8) + (i & 7) + ((i >> 3) << 4);
  float v = (k < Kreal) ? W[(size_t)n * Kreal + k] : 0.0f;
  out[idx] = (__bf16)v;
}

// Segment bounds of the sorted lg_dst array: rowptr[e] = lower_bound(lg_dst, e).
// Computed once per launch; reused by all 5 BP iterations.
__global__ __launch_bounds__(256) void mpnn_rowptr(
    const int* __restrict__ lg_dst, int L, int* __restrict__ rowptr) {
  int e = blockIdx.x * blockDim.x + threadIdx.x;
  if (e > NEDGES) return;
  if (e == NEDGES) { rowptr[e] = L; return; }
  int lo = 0, hi = L;
  while (lo < hi) { int mid = (lo + hi) >> 1; if (lg_dst[mid] < e) lo = mid + 1; else hi = mid; }
  rowptr[e] = lo;
}

// A fragment (16-bit A 16x32 layout): lane<16: row=lane, K=kk*32+{0..7,16..23};
// lane>=16: row=lane-16, K=kk*32+{8..15,24..31}. Two 16B LDS loads per lane.
__device__ __forceinline__ v16bf mpnn_load_afrag(const __bf16* a_lds, int lane,
                                                 int kk, int stride) {
  int row = lane & 15;
  int kbase = kk * 32 + ((lane < 16) ? 0 : 8);
  const __bf16* p = a_lds + row * stride + kbase;
  union { v16bf v; v8bf h[2]; } u;
  u.h[0] = *(const v8bf*)(p);
  u.h[1] = *(const v8bf*)(p + 16);
  return u.v;
}

__device__ __forceinline__ v16bf mpnn_load_bfrag(const __bf16* wfrag, int nt,
                                                 int kk, int nkk, int lane) {
  const __bf16* p = wfrag + (((size_t)(nt * nkk + kk) * 32 + lane) << 4);
  return *(const v16bf*)p;
}

__device__ __forceinline__ v8f mpnn_wmma(v16bf a, v16bf b, v8f c) {
  return __builtin_amdgcn_wmma_f32_16x16x32_bf16(false, a, false, b,
                                                 (short)0, c, false, false);
}

// ---------------------------------------------------------------------------
// GEMM0: msg_input = concat(x[src],bond) @ W_i^T ; msg = relu(msg_input)
// One block = 16 edges x 256 outputs. K padded 50 -> 64 (nkk = 2).
// ---------------------------------------------------------------------------
#define KP0   64
#define APAD0 72
__global__ __launch_bounds__(256) void mpnn_gemm_init(
    const float* __restrict__ x, const float* __restrict__ bond,
    const int* __restrict__ edge_src, const __bf16* __restrict__ wif,
    float* __restrict__ msg_input, float* __restrict__ msg) {
  __shared__ __bf16 a_lds[16 * APAD0];
  int r0 = blockIdx.x * 16;
  int t = threadIdx.x;
  {
    int row = t >> 4;
    int e = r0 + row;
    int s = edge_src[e];
    for (int k = (t & 15); k < KP0; k += 16) {
      float v = 0.0f;
      if (k < AF)            v = x[(size_t)s * AF + k];
      else if (k < AF + BFD) v = bond[(size_t)e * BFD + (k - AF)];
      a_lds[row * APAD0 + k] = (__bf16)v;
    }
  }
  __syncthreads();
  int wave = t >> 5, lane = t & 31;
  int nt0 = wave * 2, nt1 = nt0 + 1;
  v8f c0 = {}; v8f c1 = {};
#pragma unroll
  for (int kk = 0; kk < 2; ++kk) {
    v16bf a  = mpnn_load_afrag(a_lds, lane, kk, APAD0);
    v16bf b0 = mpnn_load_bfrag(wif, nt0, kk, 2, lane);
    v16bf b1 = mpnn_load_bfrag(wif, nt1, kk, 2, lane);
    c0 = mpnn_wmma(a, b0, c0);
    c1 = mpnn_wmma(a, b1, c1);
  }
  int col0 = nt0 * 16 + (lane & 15);
  int rowadd = (lane < 16) ? 0 : 8;
#pragma unroll
  for (int r = 0; r < 8; ++r) {
    size_t row = (size_t)(r0 + r + rowadd);
    size_t i0 = row * H + col0;
    size_t i1 = i0 + 16;
    float v0 = c0[r], v1 = c1[r];
    msg_input[i0] = v0;  msg[i0] = v0 > 0.0f ? v0 : 0.0f;
    msg_input[i1] = v1;  msg[i1] = v1 > 0.0f ? v1 : 0.0f;
  }
}

// ---------------------------------------------------------------------------
// Fused hot step (x5): for each edge tile, gather+sum incoming line-graph
// messages (f32) directly into the bf16 A tile in LDS, then
// msg_out = relu(msg_input + gathered @ W_h^T).  K = 256 (nkk = 8).
// msg_in/msg_out ping-pong to avoid cross-block RAW on the gather.
// ---------------------------------------------------------------------------
#define APAD1 264
__global__ __launch_bounds__(256) void mpnn_gemm_h_fused(
    const float* __restrict__ msg_in, const float* __restrict__ msg_input,
    const int* __restrict__ lg_src, const int* __restrict__ rowptr,
    const __bf16* __restrict__ whf, float* __restrict__ msg_out) {
  __shared__ __bf16 a_lds[16 * APAD1];
  int r0 = blockIdx.x * 16;
  int t = threadIdx.x;
  {
    int row = t >> 4, k0 = (t & 15) << 4;
    int e2 = r0 + row;
    int start = rowptr[e2];
    int end   = rowptr[e2 + 1];
    float acc[16];
#pragma unroll
    for (int i = 0; i < 16; ++i) acc[i] = 0.0f;
    for (int j = start; j < end; ++j) {
      const float4* src = (const float4*)(msg_in + (size_t)lg_src[j] * H + k0);
#pragma unroll
      for (int q = 0; q < 4; ++q) {
        float4 f = src[q];
        acc[q * 4 + 0] += f.x; acc[q * 4 + 1] += f.y;
        acc[q * 4 + 2] += f.z; acc[q * 4 + 3] += f.w;
      }
    }
    __bf16* dst = a_lds + row * APAD1 + k0;
#pragma unroll
    for (int i = 0; i < 16; ++i) dst[i] = (__bf16)acc[i];
  }
  __syncthreads();
  int wave = t >> 5, lane = t & 31;
  int nt0 = wave * 2, nt1 = nt0 + 1;
  v8f c0 = {}; v8f c1 = {};
#pragma unroll
  for (int kk = 0; kk < 8; ++kk) {
    v16bf a  = mpnn_load_afrag(a_lds, lane, kk, APAD1);
    v16bf b0 = mpnn_load_bfrag(whf, nt0, kk, 8, lane);
    v16bf b1 = mpnn_load_bfrag(whf, nt1, kk, 8, lane);
    c0 = mpnn_wmma(a, b0, c0);
    c1 = mpnn_wmma(a, b1, c1);
  }
  int col0 = nt0 * 16 + (lane & 15);
  int rowadd = (lane < 16) ? 0 : 8;
#pragma unroll
  for (int r = 0; r < 8; ++r) {
    size_t row = (size_t)(r0 + r + rowadd);
    size_t i0 = row * H + col0;
    size_t i1 = i0 + 16;
    float v0 = msg_input[i0] + c0[r];
    float v1 = msg_input[i1] + c1[r];
    msg_out[i0] = v0 > 0.0f ? v0 : 0.0f;
    msg_out[i1] = v1 > 0.0f ? v1 : 0.0f;
  }
}

// ---------------------------------------------------------------------------
// GEMM2: h = relu(concat(x, m) @ W_o^T + b_o). K padded 295 -> 320 (nkk = 10).
// ---------------------------------------------------------------------------
#define KP2   320
#define APAD2 328
__global__ __launch_bounds__(256) void mpnn_gemm_out(
    const float* __restrict__ x, const float* __restrict__ m,
    const __bf16* __restrict__ wof, const float* __restrict__ b_o,
    float* __restrict__ h) {
  __shared__ __bf16 a_lds[16 * APAD2];
  int r0 = blockIdx.x * 16;
  int t = threadIdx.x;
  {
    int row = t >> 4;
    int node = r0 + row;
    for (int k = (t & 15); k < KP2; k += 16) {
      float v = 0.0f;
      if (k < AF)          v = x[(size_t)node * AF + k];
      else if (k < AF + H) v = m[(size_t)node * H + (k - AF)];
      a_lds[row * APAD2 + k] = (__bf16)v;
    }
  }
  __syncthreads();
  int wave = t >> 5, lane = t & 31;
  int nt0 = wave * 2, nt1 = nt0 + 1;
  v8f c0 = {}; v8f c1 = {};
#pragma unroll
  for (int kk = 0; kk < 10; ++kk) {
    v16bf a  = mpnn_load_afrag(a_lds, lane, kk, APAD2);
    v16bf b0 = mpnn_load_bfrag(wof, nt0, kk, 10, lane);
    v16bf b1 = mpnn_load_bfrag(wof, nt1, kk, 10, lane);
    c0 = mpnn_wmma(a, b0, c0);
    c1 = mpnn_wmma(a, b1, c1);
  }
  int col0 = nt0 * 16 + (lane & 15);
  float bo0 = b_o[col0];
  float bo1 = b_o[col0 + 16];
  int rowadd = (lane < 16) ? 0 : 8;
#pragma unroll
  for (int r = 0; r < 8; ++r) {
    size_t row = (size_t)(r0 + r + rowadd);
    size_t i0 = row * H + col0;
    size_t i1 = i0 + 16;
    float v0 = c0[r] + bo0;
    float v1 = c1[r] + bo1;
    h[i0] = v0 > 0.0f ? v0 : 0.0f;
    h[i1] = v1 > 0.0f ? v1 : 0.0f;
  }
}

// m[edge_dst[e], :] += msg[e, :]  (edge_dst unsorted -> f32 atomics)
__global__ __launch_bounds__(256) void mpnn_node_scatter(
    const float* __restrict__ msg, const int* __restrict__ edge_dst,
    float* __restrict__ m) {
  int gid = blockIdx.x * blockDim.x + threadIdx.x;
  int e = gid >> 6;
  int part = gid & 63;
  if (e >= NEDGES) return;
  int d = edge_dst[e];
  float4 v = ((const float4*)(msg + (size_t)e * H))[part];
  float* base = m + (size_t)d * H + part * 4;
  atomicAdd(base + 0, v.x);
  atomicAdd(base + 1, v.y);
  atomicAdd(base + 2, v.z);
  atomicAdd(base + 3, v.w);
}

// Mean readout: exactly APM atoms per molecule by construction.
__global__ __launch_bounds__(256) void mpnn_readout(
    const float* __restrict__ h, float* __restrict__ out) {
  int gid = blockIdx.x * blockDim.x + threadIdx.x;
  if (gid >= NMOLS * H) return;
  int g = gid >> 8;
  int c = gid & 255;
  const float* base = h + (size_t)g * APM * H + c;
  float s = 0.0f;
#pragma unroll
  for (int a = 0; a < APM; ++a) s += base[(size_t)a * H];
  out[gid] = s * (1.0f / (float)APM);
}

extern "C" void kernel_launch(void* const* d_in, const int* in_sizes, int n_in,
                              void* d_out, int out_size, void* d_ws, size_t ws_size,
                              hipStream_t stream) {
  (void)n_in; (void)out_size; (void)ws_size;
  const float* x        = (const float*)d_in[0];
  const float* bond_x   = (const float*)d_in[1];
  const float* W_i      = (const float*)d_in[2];
  const float* W_h      = (const float*)d_in[3];
  const float* W_o      = (const float*)d_in[4];
  const float* b_o      = (const float*)d_in[5];
  const int*   edge_src = (const int*)d_in[6];
  const int*   edge_dst = (const int*)d_in[7];
  const int*   lg_src   = (const int*)d_in[8];
  const int*   lg_dst   = (const int*)d_in[9];
  int L = in_sizes[8];

  char* ws = (char*)d_ws;
  size_t szEH = (size_t)NEDGES * H * sizeof(float);
  float*  msg_input = (float*)(ws);
  float*  msgA      = (float*)(ws + szEH);
  float*  msgB      = (float*)(ws + 2 * szEH);
  __bf16* wif       = (__bf16*)(ws + 3 * szEH);
  __bf16* whf       = wif + 16 * 2  * 32 * 16;
  __bf16* wof       = whf + 16 * 8  * 32 * 16;
  int*    rowptr    = (int*)(wof + 16 * 10 * 32 * 16);

  // One-time per launch: pack weights to fragment-native bf16, build segment
  // bounds for the (sorted) line-graph destination list.
  mpnn_pack_w<<<(16 * 2  * 512 + 255) / 256, 256, 0, stream>>>(W_i, AF + BFD, 2,  wif);
  mpnn_pack_w<<<(16 * 8  * 512 + 255) / 256, 256, 0, stream>>>(W_h, H,        8,  whf);
  mpnn_pack_w<<<(16 * 10 * 512 + 255) / 256, 256, 0, stream>>>(W_o, AF + H,   10, wof);
  mpnn_rowptr<<<(NEDGES + 1 + 255) / 256, 256, 0, stream>>>(lg_dst, L, rowptr);

  const int tilesE = NEDGES / 16;   // 23750, exact
  const int tilesN = NNODES / 16;   // 12500, exact

  mpnn_gemm_init<<<tilesE, 256, 0, stream>>>(x, bond_x, edge_src, wif, msg_input, msgA);

  // DEPTH-1 = 5 fused gather+GEMM iterations, ping-pong A<->B (ends in msgB).
  float* cur = msgA;
  float* nxt = msgB;
  for (int it = 0; it < DEPTH - 1; ++it) {
    mpnn_gemm_h_fused<<<tilesE, 256, 0, stream>>>(cur, msg_input, lg_src, rowptr,
                                                  whf, nxt);
    float* tmp = cur; cur = nxt; nxt = tmp;
  }
  float* msg_final = cur;  // = msgB after 5 iterations

  float* m = nxt;  // the other ping-pong buffer (msgA); N*H <= E*H
  hipMemsetAsync(m, 0, (size_t)NNODES * H * sizeof(float), stream);
  mpnn_node_scatter<<<((size_t)NEDGES * 64 + 255) / 256, 256, 0, stream>>>(
      msg_final, edge_dst, m);

  float* h = msg_input;  // reuse: msg_input no longer needed
  mpnn_gemm_out<<<tilesN, 256, 0, stream>>>(x, m, wof, b_o, h);
  mpnn_readout<<<(NMOLS * H + 255) / 256, 256, 0, stream>>>(h, (float*)d_out);
}
